// DMCNN_74603581932108
// MI455X (gfx1250) — matrix operand order repaired and checked by reading, not compiled, not run on hardware
//
#include <hip/hip_runtime.h>
#include <hip/hip_bf16.h>
#include <stdint.h>

// ---- problem constants ----
#define E_DIM 300
#define EPAD  320          // K-pad for conv_tok GEMM (10 x 32)
#define P_DIM 5
#define H_DIM 200
#define S_DIM 256
#define SH    (S_DIM + 2)  // halo-padded rows per batch (s = -1 .. 256)
#define L_DIM 34
#define B_DIM 8
#define NROWS (B_DIM * S_DIM)        // 2048 GEMM1 output rows
#define HROWS (B_DIM * SH)           // 2064 halo-padded tok rows
#define NT1   14                     // 224 cols (H padded, even #tiles) / 16
#define NP1   (NT1 / 2)              // 7 n-tile pairs per wave
#define KS1   10                     // 320/32 K-steps per conv tap
#define FROWS (B_DIM * (S_DIM - 2))  // 2032 FC rows
#define FK    1300
#define FKPAD 1312                   // 41 x 32
#define KS2   41
#define NT2   3                      // 48 cols (L padded) / 16

typedef __attribute__((ext_vector_type(16))) __bf16    v16bf;
typedef __attribute__((ext_vector_type(8)))  float     v8f;
typedef __attribute__((ext_vector_type(8)))  uint32_t  v8u;

union BF16Frag { uint4 q[2]; v8u u8; v16bf v; };

__device__ __forceinline__ unsigned short f2bf(float f) {
  uint32_t u = __builtin_bit_cast(uint32_t, f);
  u += 0x7FFFu + ((u >> 16) & 1u);               // round-to-nearest-even
  return (unsigned short)(u >> 16);
}

// ---------------- kernel 1: embedding gather -> bf16, K-padded + halo rows ---
// tok[b][s] lives at row (b*SH + s + 1); rows s=-1 and s=256 are zero so the
// conv GEMM needs no boundary branches.
__global__ void k_gather(const int* __restrict__ inputs,
                         const float* __restrict__ word_emb,
                         unsigned short* __restrict__ tokb) {
  int idx = blockIdx.x * blockDim.x + threadIdx.x;
  if (idx >= HROWS * EPAD) return;
  int row = idx / EPAD, e = idx - row * EPAD;
  int b = row / SH, sr = row - b * SH;
  int s = sr - 1;                                // -1 .. 256
  float v = 0.f;
  if (s >= 0 && s < S_DIM && e < E_DIM)
    v = word_emb[(size_t)inputs[b * S_DIM + s] * E_DIM + e];
  tokb[idx] = f2bf(v);
}

// ---------------- kernel 2: pack weights into WMMA B-fragment order ----------
// B fragment order used here (16-bit, 32x16; A loads use the same k order): lane = {half = lane>>4, n = lane&15},
// element e in [0,16): K = half*16 + e  -> 16 contiguous bf16 per lane.
__global__ void k_pack(const float* __restrict__ conv_W,
                       const float* __restrict__ fc_W,
                       unsigned short* __restrict__ wtokp,
                       unsigned short* __restrict__ fcwp) {
  int idx = blockIdx.x * blockDim.x + threadIdx.x;
  const int N1 = 3 * KS1 * NT1 * 16 * 32;   // conv_tok B (incl. zero tile 13)
  const int N2 = KS2 * NT2 * 16 * 32;       // fc B
  if (idx < N1) {
    int t = idx;
    int e    = t & 15; t >>= 4;
    int half = t & 1;  t >>= 1;
    int nl   = t & 15; t >>= 4;
    int tn   = t % NT1; t /= NT1;
    int k    = t / KS1, kk = t % KS1;       // tap, K-step
    int kd   = kk * 32 + half * 16 + e;     // contraction index (E dim)
    int h    = tn * 16 + nl;
    float v = 0.f;
    if (kd < E_DIM && h < H_DIM)
      v = conv_W[(size_t)h * (E_DIM + P_DIM) * 3 + (size_t)kd * 3 + k];
    wtokp[idx] = f2bf(v);
  } else if (idx < N1 + N2) {
    int t = idx - N1;
    int e    = t & 15; t >>= 4;
    int half = t & 1;  t >>= 1;
    int nl   = t & 15; t >>= 4;
    int tn   = t % NT2; t /= NT2;
    int kk   = t;
    int f    = kk * 32 + half * 16 + e;
    int l    = tn * 16 + nl;
    float v = 0.f;
    if (f < FK && l < L_DIM) v = fc_W[(size_t)l * FK + f];
    fcwp[idx] = f2bf(v);
  }
}

// ---------------- kernel 3: conv_tok as 3-tap WMMA GEMM ----------------------
// Ctok[(b,s), h] = bias[h] + sum_k sum_e tok[b, s+k-1, e] * W_tok[h, e, k]
// One wave computes a 16x32 output tile (2 n-tiles share each A fragment).
// Halo buffer makes the inner loop branch-free: 2 A loads + 2 B loads + 2 WMMA.
__global__ void k_conv_tok(const unsigned short* __restrict__ tokb,
                           const unsigned short* __restrict__ wtokp,
                           const float* __restrict__ conv_b,
                           float* __restrict__ Ctok) {
  int wave = threadIdx.x >> 5;
  int lane = threadIdx.x & 31;
  int tile = blockIdx.x * (blockDim.x >> 5) + wave;
  const int MT = NROWS / 16;                 // 128
  if (tile >= MT * NP1) return;              // wave-uniform guard (EXEC stays full)
  int tm = tile / NP1, tp = tile - tm * NP1;
  int tn0 = tp * 2, tn1 = tn0 + 1;
  int half = lane >> 4, nl = lane & 15;
  int base_row = tm * 16;
  int b = base_row >> 8;
  int s = (base_row & 255) + nl;             // this lane's A row (M = nl)
  v8f acc0 = {}, acc1 = {};
  for (int k = 0; k < 3; ++k) {
    const unsigned short* arow = tokb + (size_t)(b * SH + s + k) * EPAD;  // s+k-1 (+1 halo)
    if (k < 2)
      __builtin_prefetch(tokb + (size_t)(b * SH + s + k + 1) * EPAD, 0, 1);
    for (int kb = 0; kb < KS1; ++kb) {
      BF16Frag a;
      a.q[0] = *(const uint4*)(arow + kb * 32 + half * 16);
      a.q[1] = *(const uint4*)(arow + kb * 32 + half * 16 + 8);
      int step = k * KS1 + kb;
      const unsigned short* bbase =
          wtokp + (((size_t)step * NT1 + tn0) * 16 + nl) * 32 + half * 16;
      BF16Frag b0, b1;
      b0.u8 = *(const v8u*)(bbase);
      b1.u8 = *(const v8u*)(bbase + 16 * 32);            // next n-tile
      acc0 = __builtin_amdgcn_wmma_f32_16x16x32_bf16(false, a.v, false, b0.v,
                                                     (short)0, acc0, false, false);
      acc1 = __builtin_amdgcn_wmma_f32_16x16x32_bf16(false, a.v, false, b1.v,
                                                     (short)0, acc1, false, false);
    }
  }
  int h0 = tn0 * 16 + nl, h1 = tn1 * 16 + nl;
  if (h0 < H_DIM) {
    float bias = conv_b[h0];
    for (int r = 0; r < 8; ++r) {
      int row = base_row + r + half * 8;     // D layout: M = r + 8*half, N = lane&15
      Ctok[(size_t)row * H_DIM + h0] = acc0[r] + bias;
    }
  }
  if (h1 < H_DIM) {
    float bias = conv_b[h1];
    for (int r = 0; r < 8; ++r) {
      int row = base_row + r + half * 8;
      Ctok[(size_t)row * H_DIM + h1] = acc1[r] + bias;
    }
  }
}

// ---------------- kernel 4: llf (3-word window) -> feat cols [400,1312) ------
__global__ void k_llf(const unsigned short* __restrict__ tokb,
                      unsigned short* __restrict__ featb) {
  const int COLS = FKPAD - 2 * H_DIM;        // 912 (900 llf + 12 zero K-pad)
  int idx = blockIdx.x * blockDim.x + threadIdx.x;
  if (idx >= FROWS * COLS) return;
  int r = idx / COLS, c = idx - r * COLS;
  int b = r / (S_DIM - 2), i = r - b * (S_DIM - 2);
  unsigned short v = 0;
  if (c < 3 * E_DIM) {
    int seg = c / E_DIM, e = c - seg * E_DIM;
    v = tokb[(size_t)(b * SH + i + seg + 1) * EPAD + e];  // tok[b, i+seg, e]
  }
  featb[(size_t)r * FKPAD + 2 * H_DIM + c] = v;
}

// ---------------- kernel 5: fused conv_pf + dynamic max-pool -----------------
// One block per pivot i (254 blocks). conv_pf (K=15) computed with VALU from
// LDS-cached pf_emb / W_pf, then left/right masked maxes vs Ctok, relu at end.
#define HCH 32
__global__ void k_pool(const float* __restrict__ Ctok,
                       const float* __restrict__ pf_emb,
                       const float* __restrict__ conv_W,
                       unsigned short* __restrict__ featb) {
  __shared__ float pfe[S_DIM * P_DIM];   // 1280 f
  __shared__ float wpf[HCH * 15];        // 480 f
  __shared__ float cpf[HCH * 257];       // 8224 f (stride 257: conflict-free)
  int i = blockIdx.x;                    // 0..253
  int ival = i + 1;
  int t = threadIdx.x;                   // 256 threads = 8 waves
  for (int idx = t; idx < S_DIM * P_DIM; idx += 256) pfe[idx] = pf_emb[idx];
  for (int h0 = 0; h0 < H_DIM; h0 += HCH) {
    int hcnt = (H_DIM - h0 < HCH) ? (H_DIM - h0) : HCH;
    __syncthreads();                     // previous chunk's readers done
    for (int idx = t; idx < hcnt * 15; idx += 256) {
      int hl = idx / 15, rem = idx - hl * 15;          // rem = p*3+k
      int p = rem / 3, k = rem - p * 3;
      wpf[idx] = conv_W[(size_t)(h0 + hl) * (E_DIM + P_DIM) * 3 + (size_t)(E_DIM + p) * 3 + k];
    }
    __syncthreads();
    // phase A: conv_pf[i, h0+hl, j] -> LDS
    for (int idx = t; idx < hcnt * S_DIM; idx += 256) {
      int hl = idx >> 8, j = idx & 255;
      float acc = 0.f;
      for (int k = 0; k < 3; ++k) {
        int jj = j + k - 1;
        if (jj < 0 || jj >= S_DIM) continue;           // conv zero-padding
        int d = jj - ival; d = (d < 0) ? -d : d;       // pf_emb[|j'-i|]
        const float* pr = &pfe[d * P_DIM];
        const float* wr = &wpf[hl * 15 + k];
        acc += wr[0] * pr[0] + wr[3] * pr[1] + wr[6] * pr[2]
             + wr[9] * pr[3] + wr[12] * pr[4];
      }
      cpf[hl * 257 + j] = acc;
    }
    __syncthreads();
    // phase B: lanes = consecutive h (coalesced Ctok), thread = (h, b) output
    int hl = t & 31, b = t >> 5;
    if (hl < hcnt) {
      int h = h0 + hl;
      const float* crow = Ctok + (size_t)b * S_DIM * H_DIM + h;
      const float* prow = cpf + hl * 257;
      float ml = -3.4e38f, mr = -3.4e38f;
      for (int j = 0; j < ival; ++j)
        ml = fmaxf(ml, crow[(size_t)j * H_DIM] + prow[j]);
      for (int j = ival; j < S_DIM; ++j)
        mr = fmaxf(mr, crow[(size_t)j * H_DIM] + prow[j]);
      size_t frow = (size_t)(b * (S_DIM - 2) + i) * FKPAD;
      featb[frow + h]         = f2bf(fmaxf(ml, 0.f));  // max(relu(x)) = relu(max(x))
      featb[frow + H_DIM + h] = f2bf(fmaxf(mr, 0.f));
    }
  }
}

// ---------------- kernel 6: final FC as WMMA GEMM + bias ---------------------
// One wave per 16-row strip computes all 3 n-tiles (48 cols, 34 valid):
// each A fragment feeds 3 WMMAs.
__global__ void k_fc(const unsigned short* __restrict__ featb,
                     const unsigned short* __restrict__ fcwp,
                     const float* __restrict__ fc_b,
                     float* __restrict__ out) {
  int wave = threadIdx.x >> 5, lane = threadIdx.x & 31;
  int tm = blockIdx.x * (blockDim.x >> 5) + wave;
  const int MT = FROWS / 16;               // 127
  if (tm >= MT) return;                    // wave-uniform
  int half = lane >> 4, nl = lane & 15;
  const unsigned short* arow = featb + (size_t)(tm * 16 + nl) * FKPAD;
  v8f acc0 = {}, acc1 = {}, acc2 = {};
  for (int kk = 0; kk < KS2; ++kk) {
    BF16Frag a;
    a.q[0] = *(const uint4*)(arow + kk * 32 + half * 16);
    a.q[1] = *(const uint4*)(arow + kk * 32 + half * 16 + 8);
    const unsigned short* bbase =
        fcwp + (((size_t)kk * NT2) * 16 + nl) * 32 + half * 16;
    BF16Frag b0, b1, b2;
    b0.u8 = *(const v8u*)(bbase);
    b1.u8 = *(const v8u*)(bbase + 16 * 32);
    b2.u8 = *(const v8u*)(bbase + 32 * 32);
    acc0 = __builtin_amdgcn_wmma_f32_16x16x32_bf16(false, a.v, false, b0.v,
                                                   (short)0, acc0, false, false);
    acc1 = __builtin_amdgcn_wmma_f32_16x16x32_bf16(false, a.v, false, b1.v,
                                                   (short)0, acc1, false, false);
    acc2 = __builtin_amdgcn_wmma_f32_16x16x32_bf16(false, a.v, false, b2.v,
                                                   (short)0, acc2, false, false);
  }
  for (int tn = 0; tn < NT2; ++tn) {
    const v8f& acc = (tn == 0) ? acc0 : (tn == 1) ? acc1 : acc2;
    int l = tn * 16 + nl;
    if (l < L_DIM) {
      float bias = fc_b[l];
      for (int r = 0; r < 8; ++r) {
        int row = tm * 16 + r + half * 8;
        int b = row / (S_DIM - 2), i = row - b * (S_DIM - 2);
        out[(size_t)(b * S_DIM + i + 1) * L_DIM + l] = acc[r] + bias;  // rows 1..254
      }
    }
  }
}

// ---------------- kernel 7: pad rows s=0 and s=255 ---------------------------
__global__ void k_pad(float* __restrict__ out) {
  int idx = blockIdx.x * blockDim.x + threadIdx.x;
  if (idx >= B_DIM * 2 * L_DIM) return;
  int l = idx % L_DIM;
  int t = idx / L_DIM;
  int b = t >> 1, side = t & 1;
  int s = side ? (S_DIM - 1) : 0;
  out[(size_t)(b * S_DIM + s) * L_DIM + l] = (l == L_DIM - 1) ? 1.0f : 0.0f;
}

extern "C" void kernel_launch(void* const* d_in, const int* in_sizes, int n_in,
                              void* d_out, int out_size, void* d_ws, size_t ws_size,
                              hipStream_t stream) {
  const int*   inputs   = (const int*)d_in[0];
  const float* word_emb = (const float*)d_in[1];
  const float* pf_emb   = (const float*)d_in[2];
  const float* conv_W   = (const float*)d_in[3];
  const float* conv_b   = (const float*)d_in[4];
  const float* fc_W     = (const float*)d_in[5];
  const float* fc_b     = (const float*)d_in[6];
  float* out = (float*)d_out;

  // workspace carve-out (~9 MB total), 256B aligned slabs
  char* ws = (char*)d_ws;
  size_t off = 0;
  auto carve = [&](size_t bytes) {
    void* p = ws + off;
    off = (off + bytes + 255) & ~(size_t)255;
    return p;
  };
  const int N1 = 3 * KS1 * NT1 * 16 * 32;                       // 215,040
  const int N2 = KS2 * NT2 * 16 * 32;                           // 62,976
  unsigned short* tokb  = (unsigned short*)carve((size_t)HROWS * EPAD * 2);
  unsigned short* wtokp = (unsigned short*)carve((size_t)N1 * 2);
  unsigned short* fcwp  = (unsigned short*)carve((size_t)N2 * 2);
  float*          Ctok  = (float*)carve((size_t)NROWS * H_DIM * 4);
  unsigned short* featb = (unsigned short*)carve((size_t)FROWS * FKPAD * 2);

  k_pack<<<(N1 + N2 + 255) / 256, 256, 0, stream>>>(conv_W, fc_W, wtokp, fcwp);
  k_gather<<<(HROWS * EPAD + 255) / 256, 256, 0, stream>>>(inputs, word_emb, tokb);
  k_conv_tok<<<(128 * NP1) / 4, 128, 0, stream>>>(tokb, wtokp, conv_b, Ctok);
  k_llf<<<(FROWS * (FKPAD - 2 * H_DIM) + 255) / 256, 256, 0, stream>>>(tokb, featb);
  k_pool<<<S_DIM - 2, 256, 0, stream>>>(Ctok, pf_emb, conv_W, featb);
  k_fc<<<(127 + 3) / 4, 128, 0, stream>>>(featb, fcwp, fc_b, out);
  k_pad<<<(B_DIM * 2 * L_DIM + 255) / 256, 256, 0, stream>>>(out);
}
